// DeformableTransformerDecoderLayer_88467736363084
// MI455X (gfx1250) — matrix operand order repaired and checked by reading, not compiled, not run on hardware
//
#include <hip/hip_runtime.h>
#include <hip/hip_bf16.h>
#include <math.h>

#define D 256
#define NH 8
#define NL 4
#define NP 4
#define DH 32
#define DFFN 1024
#define NQ 900
#define BS 4
#define S_TOT 11253
#define MTOK (NQ*BS)      // 3600
#define MMEM (S_TOT*BS)   // 45012

typedef __attribute__((ext_vector_type(16))) _Float16 v16h;
typedef __attribute__((ext_vector_type(8)))  float    v8f;
typedef __attribute__((ext_vector_type(4)))  _Float16 h4;

__device__ __forceinline__ h4 cvt4(float4 v) {
    h4 r;
    r[0] = (_Float16)v.x; r[1] = (_Float16)v.y;
    r[2] = (_Float16)v.z; r[3] = (_Float16)v.w;
    return r;
}

// ---------------------------------------------------------------------------
// Elementwise add: o = a + b
// ---------------------------------------------------------------------------
__global__ __launch_bounds__(256)
void ew_add(const float* __restrict__ a, const float* __restrict__ b,
            float* __restrict__ o, int n) {
    int i = blockIdx.x * 256 + threadIdx.x;
    if (i < n) o[i] = a[i] + b[i];
}

// ---------------------------------------------------------------------------
// Generic WMMA GEMM: C[m*ldc+coff+n] = act( sum_k A[m*lda+k]*W[n*ldw+k] + bias[n] )
// 128 threads = 4 waves; tile 64x64, K step 32; 2x2 v_wmma_f32_16x16x32_f16
// tiles per wave. Staging: batched global_load_b128, then cvt + ds_store_b64.
// ---------------------------------------------------------------------------
__device__ __forceinline__ void gemm_store(v8f c, int m0, int n0,
                                           const float* __restrict__ bias,
                                           float* __restrict__ C, int ldc, int coff,
                                           int M, int N, int relu,
                                           int l15, int half8) {
    int n = n0 + l15;
    if (n >= N) return;
    float bv = bias ? bias[n] : 0.0f;
#pragma unroll
    for (int r = 0; r < 8; ++r) {
        int m = m0 + r + half8;           // C layout: M = vgpr + 8*(lane>=16)
        if (m < M) {
            float v = c[r] + bv;
            if (relu) v = fmaxf(v, 0.0f);
            C[(long)m * ldc + coff + n] = v;
        }
    }
}

__global__ __launch_bounds__(128)
void gemm_wmma_f16(const float* __restrict__ A, int lda,
                   const float* __restrict__ W, int ldw,
                   const float* __restrict__ bias,
                   float* __restrict__ C, int ldc, int coff,
                   int M, int N, int K, int relu) {
    __shared__ _Float16 As[64][36];
    __shared__ _Float16 Bs[64][36];

    const int lane  = threadIdx.x & 31;
    const int wave  = threadIdx.x >> 5;
    const int wm    = (wave >> 1) * 32;
    const int wn    = (wave & 1) * 32;
    const int bm    = blockIdx.y * 64;
    const int bn    = blockIdx.x * 64;
    const int l15   = lane & 15;
    const int half8 = (lane >> 4) * 8;    // M offset for C
    const int half16= (lane >> 4) * 16;   // K offset for B-frag

    v8f acc00 = {}, acc01 = {}, acc10 = {}, acc11 = {};

    for (int k0 = 0; k0 < K; k0 += 32) {
        // Phase 1: issue all 8 b128 loads (A tile: 4, W tile: 4)
        float4 va[4], vb[4];
        bool   ma[4], mb[4];
#pragma unroll
        for (int t = 0; t < 4; ++t) {
            int i  = threadIdx.x + t * 128;
            int r  = i >> 3, c4 = (i & 7) * 4;
            int gm = bm + r;
            ma[t]  = (gm < M);
            int gmc = ma[t] ? gm : (M - 1);
            va[t]  = *(const float4*)(A + (long)gmc * lda + k0 + c4);
            int gn = bn + r;
            mb[t]  = (gn < N);
            int gnc = mb[t] ? gn : (N - 1);
            vb[t]  = *(const float4*)(W + (long)gnc * ldw + k0 + c4);
        }
        // CDNA5 prefetch of the next A tile (global_prefetch_b8)
        if (k0 + 32 < K) {
            int pr = bm + (threadIdx.x >> 1);
            if (pr < M)
                __builtin_prefetch(&A[(long)pr * lda + k0 + 32 + (threadIdx.x & 1) * 16], 0, 1);
        }
        // Phase 2: convert + store to LDS
#pragma unroll
        for (int t = 0; t < 4; ++t) {
            int i  = threadIdx.x + t * 128;
            int r  = i >> 3, c4 = (i & 7) * 4;
            float4 a = ma[t] ? va[t] : make_float4(0.f, 0.f, 0.f, 0.f);
            float4 bb = mb[t] ? vb[t] : make_float4(0.f, 0.f, 0.f, 0.f);
            *(h4*)&As[r][c4] = cvt4(a);
            *(h4*)&Bs[r][c4] = cvt4(bb);
        }
        __syncthreads();

        // Build fragments: A and B both use contiguous k = 16*half + s
        v16h a0, a1, b0, b1;
        {
            int m0r = wm + l15, m1r = wm + 16 + l15;
#pragma unroll
            for (int i = 0; i < 8; ++i) {
                int kb = half16 + i * 2;
                a0[2*i]   = As[m0r][kb];
                a0[2*i+1] = As[m0r][kb + 1];
                a1[2*i]   = As[m1r][kb];
                a1[2*i+1] = As[m1r][kb + 1];
            }
            int n0r = wn + l15, n1r = wn + 16 + l15;
#pragma unroll
            for (int j = 0; j < 16; ++j) {
                b0[j] = Bs[n0r][half16 + j];
                b1[j] = Bs[n1r][half16 + j];
            }
        }
        acc00 = __builtin_amdgcn_wmma_f32_16x16x32_f16(false, a0, false, b0, (short)0, acc00, false, false);
        acc01 = __builtin_amdgcn_wmma_f32_16x16x32_f16(false, a0, false, b1, (short)0, acc01, false, false);
        acc10 = __builtin_amdgcn_wmma_f32_16x16x32_f16(false, a1, false, b0, (short)0, acc10, false, false);
        acc11 = __builtin_amdgcn_wmma_f32_16x16x32_f16(false, a1, false, b1, (short)0, acc11, false, false);
        __syncthreads();
    }

    gemm_store(acc00, bm + wm,      bn + wn,      bias, C, ldc, coff, M, N, relu, l15, half8);
    gemm_store(acc01, bm + wm,      bn + wn + 16, bias, C, ldc, coff, M, N, relu, l15, half8);
    gemm_store(acc10, bm + wm + 16, bn + wn,      bias, C, ldc, coff, M, N, relu, l15, half8);
    gemm_store(acc11, bm + wm + 16, bn + wn + 16, bias, C, ldc, coff, M, N, relu, l15, half8);
}

// ---------------------------------------------------------------------------
// Self-attention (flash style): one wave handles 16 queries for one (b,h).
// qkv layout: row (q*BS+b), 3*D columns: [Q | K | V], head slice h*DH.
// Q/K/V chunks staged through LDS with batched b128 loads; WMMA for QK^T, PV.
// ---------------------------------------------------------------------------
__global__ __launch_bounds__(32)
void attn_flash(const float* __restrict__ qkv, float* __restrict__ attn) {
    __shared__ _Float16 P[16][928];       // probabilities, padded to 29 x 32
    __shared__ _Float16 Qs[16][36];
    __shared__ _Float16 Ks[16][36];
    __shared__ _Float16 Vs[32][36];
    const int lane = threadIdx.x;
    const int qb   = blockIdx.x * 16;
    const int h    = blockIdx.y, b = blockIdx.z;
    const int l15  = lane & 15;
    const int hi   = lane >> 4;

    // Stage Q block (16 x 32) once: 128 float4s, 4 per lane (batched loads)
    {
        float4 v[4];
#pragma unroll
        for (int t = 0; t < 4; ++t) {
            int i = lane + t * 32;
            int r = i >> 3, c4 = (i & 7) * 4;
            int q = qb + r; if (q >= NQ) q = NQ - 1;
            v[t] = *(const float4*)(qkv + ((long)q * BS + b) * (3 * D) + h * DH + c4);
        }
#pragma unroll
        for (int t = 0; t < 4; ++t) {
            int i = lane + t * 32;
            int r = i >> 3, c4 = (i & 7) * 4;
            *(h4*)&Qs[r][c4] = cvt4(v[t]);
        }
    }
    __syncthreads();

    v16h qa;
#pragma unroll
    for (int i = 0; i < 8; ++i) {
        int kb = hi * 16 + i * 2;
        qa[2*i]   = Qs[l15][kb];
        qa[2*i+1] = Qs[l15][kb + 1];
    }
    const float scale = 0.17677669529663689f;  // 1/sqrt(DH)

    // Pass 1: scores = Q K^T * scale -> P (LDS)
    for (int kc = 0; kc < NQ; kc += 16) {
        {
            float4 v[4];
#pragma unroll
            for (int t = 0; t < 4; ++t) {
                int i = lane + t * 32;
                int r = i >> 3, c4 = (i & 7) * 4;
                int key = kc + r; if (key >= NQ) key = NQ - 1;
                v[t] = *(const float4*)(qkv + ((long)key * BS + b) * (3 * D) + D + h * DH + c4);
            }
#pragma unroll
            for (int t = 0; t < 4; ++t) {
                int i = lane + t * 32;
                int r = i >> 3, c4 = (i & 7) * 4;
                *(h4*)&Ks[r][c4] = cvt4(v[t]);
            }
        }
        __syncthreads();
        v16h kf;
#pragma unroll
        for (int j = 0; j < 16; ++j) kf[j] = Ks[l15][hi * 16 + j];
        v8f c = {};
        c = __builtin_amdgcn_wmma_f32_16x16x32_f16(false, qa, false, kf, (short)0, c, false, false);
        int key = kc + l15;
        if (key < NQ) {
#pragma unroll
            for (int r = 0; r < 8; ++r)
                P[r + hi * 8][key] = (_Float16)(c[r] * scale);
        }
        __syncthreads();
    }

    // Softmax per query row (lanes 0..15), 4-wide over f16 rows
    if (lane < 16) {
        float mx = -1e30f;
        for (int k = 0; k < NQ; k += 4) {
            h4 v = *(h4*)&P[lane][k];
            mx = fmaxf(mx, fmaxf(fmaxf((float)v[0], (float)v[1]),
                                 fmaxf((float)v[2], (float)v[3])));
        }
        float sum = 0.0f;
        for (int k = 0; k < NQ; k += 4) {
            h4 v = *(h4*)&P[lane][k];
            h4 e;
#pragma unroll
            for (int j = 0; j < 4; ++j) {
                float ev = __expf((float)v[j] - mx);
                sum += ev;
                e[j] = (_Float16)ev;
            }
            *(h4*)&P[lane][k] = e;
        }
        float inv = 1.0f / sum;
        for (int k = 0; k < NQ; k += 4) {
            h4 v = *(h4*)&P[lane][k];
#pragma unroll
            for (int j = 0; j < 4; ++j) v[j] = (_Float16)((float)v[j] * inv);
            *(h4*)&P[lane][k] = v;
        }
        h4 z = {(_Float16)0.f, (_Float16)0.f, (_Float16)0.f, (_Float16)0.f};
        for (int k = NQ; k < 928; k += 4) *(h4*)&P[lane][k] = z;
    }
    __syncthreads();

    // Pass 2: O = P V  (two 16-wide dh tiles)
    v8f acc0 = {}, acc1 = {};
    for (int kc = 0; kc < 928; kc += 32) {
        // Stage V chunk (32 keys x 32 dh): 256 float4s, 8 per lane (batched)
        {
            float4 v[8];
#pragma unroll
            for (int t = 0; t < 8; ++t) {
                int i = lane + t * 32;
                int r = i >> 3, c4 = (i & 7) * 4;
                int key = kc + r; if (key >= NQ) key = 0;   // P zero there
                v[t] = *(const float4*)(qkv + ((long)key * BS + b) * (3 * D) + 2 * D + h * DH + c4);
            }
#pragma unroll
            for (int t = 0; t < 8; ++t) {
                int i = lane + t * 32;
                int r = i >> 3, c4 = (i & 7) * 4;
                *(h4*)&Vs[r][c4] = cvt4(v[t]);
            }
        }
        __syncthreads();
        v16h pa;
#pragma unroll
        for (int i = 0; i < 8; ++i) {
            int kb = kc + hi * 16 + i * 2;
            pa[2*i]   = P[l15][kb];
            pa[2*i+1] = P[l15][kb + 1];
        }
        v16h v0, v1;
#pragma unroll
        for (int j = 0; j < 16; ++j) {
            v0[j] = Vs[hi * 16 + j][l15];
            v1[j] = Vs[hi * 16 + j][16 + l15];
        }
        acc0 = __builtin_amdgcn_wmma_f32_16x16x32_f16(false, pa, false, v0, (short)0, acc0, false, false);
        acc1 = __builtin_amdgcn_wmma_f32_16x16x32_f16(false, pa, false, v1, (short)0, acc1, false, false);
        __syncthreads();
    }
#pragma unroll
    for (int r = 0; r < 8; ++r) {
        int q = qb + r + hi * 8;
        if (q < NQ) {
            float* op = attn + ((long)q * BS + b) * D + h * DH;
            op[l15]      = acc0[r];
            op[16 + l15] = acc1[r];
        }
    }
}

// ---------------------------------------------------------------------------
// Multi-scale deformable sampling. One block per (q,b); thread t -> (h, dh).
// ---------------------------------------------------------------------------
__global__ __launch_bounds__(256)
void msdeform_sample(const float* __restrict__ ref,    // (NQ,BS,NL,2)
                     const float* __restrict__ value,  // (S,BS,D) rows s*BS+b
                     const float* __restrict__ off,    // (NQ*BS, 256)
                     const float* __restrict__ aw,     // (NQ*BS, 128)
                     float* __restrict__ samp)         // (NQ,BS,D)
{
    const int q = blockIdx.x, b = blockIdx.y;
    const int t = threadIdx.x;
    const int h = t >> 5, dh = t & 31;
    const long row = (long)q * BS + b;
    const int Hs[4] = {92, 46, 23, 12};
    const int Wl[4] = {92, 46, 23, 12};
    const int St[4] = {0, 8464, 10580, 11109};

    // softmax over the 16 (level,point) logits of this head
    float wts[16];
    {
        const float* lg = aw + row * (NH * NL * NP) + h * (NL * NP);
        float mx = -1e30f;
#pragma unroll
        for (int i = 0; i < 16; ++i) mx = fmaxf(mx, lg[i]);
        float sum = 0.0f;
#pragma unroll
        for (int i = 0; i < 16; ++i) { wts[i] = __expf(lg[i] - mx); sum += wts[i]; }
        float inv = 1.0f / sum;
#pragma unroll
        for (int i = 0; i < 16; ++i) wts[i] *= inv;
    }

    float acc = 0.0f;
    const float* offp = off + row * (NH * NL * NP * 2) + h * (NL * NP * 2);
    for (int l = 0; l < 4; ++l) {
        const int H = Hs[l], W = Wl[l], s0 = St[l];
        const float rx = ref[(row * NL + l) * 2 + 0];
        const float ry = ref[(row * NL + l) * 2 + 1];
        for (int p = 0; p < 4; ++p) {
            float ox = offp[(l * NP + p) * 2 + 0];
            float oy = offp[(l * NP + p) * 2 + 1];
            float px = (rx + ox / (float)W) * W - 0.5f;
            float py = (ry + oy / (float)H) * H - 0.5f;
            float fx = floorf(px), fy = floorf(py);
            int   x0 = (int)fx, y0 = (int)fy;
            float wx = px - fx, wy = py - fy;
            float wc[4] = {(1.f - wx) * (1.f - wy), wx * (1.f - wy),
                           (1.f - wx) * wy,         wx * wy};
            float sv = 0.0f;
#pragma unroll
            for (int c = 0; c < 4; ++c) {
                int xi = x0 + (c & 1), yi = y0 + (c >> 1);
                if (xi >= 0 && xi < W && yi >= 0 && yi < H)
                    sv += wc[c] * value[((long)(s0 + yi * W + xi) * BS + b) * D + h * DH + dh];
            }
            acc += wts[l * 4 + p] * sv;
        }
    }
    samp[row * D + t] = acc;
}

// ---------------------------------------------------------------------------
// Fused residual + LayerNorm: out = LN(a + b) * g + beta. One row per block.
// ---------------------------------------------------------------------------
__global__ __launch_bounds__(256)
void resln(const float* __restrict__ A, const float* __restrict__ Bv,
           const float* __restrict__ g, const float* __restrict__ be,
           float* __restrict__ out) {
    __shared__ float rs[256], rq[256];
    const long row = blockIdx.x;
    const int  t   = threadIdx.x;
    float x = A[row * D + t] + Bv[row * D + t];
    rs[t] = x; rq[t] = x * x;
    __syncthreads();
    for (int s = 128; s > 0; s >>= 1) {
        if (t < s) { rs[t] += rs[t + s]; rq[t] += rq[t + s]; }
        __syncthreads();
    }
    float mean = rs[0] * (1.0f / D);
    float var  = rq[0] * (1.0f / D) - mean * mean;
    out[row * D + t] = (x - mean) * rsqrtf(var + 1e-5f) * g[t] + be[t];
}

// ---------------------------------------------------------------------------
static void launch_gemm(const float* A, int lda, const float* W, int ldw,
                        const float* bias, float* C, int ldc, int coff,
                        int M, int N, int K, int relu, hipStream_t s) {
    dim3 g((N + 63) / 64, (M + 63) / 64);
    gemm_wmma_f16<<<g, 128, 0, s>>>(A, lda, W, ldw, bias, C, ldc, coff, M, N, K, relu);
}

extern "C" void kernel_launch(void* const* d_in, const int* in_sizes, int n_in,
                              void* d_out, int out_size, void* d_ws, size_t ws_size,
                              hipStream_t stream) {
    (void)in_sizes; (void)n_in; (void)out_size; (void)ws_size;

    const float* tgt    = (const float*)d_in[0];
    const float* pos    = (const float*)d_in[1];
    const float* ref    = (const float*)d_in[2];
    const float* memory = (const float*)d_in[3];
    // d_in[4], d_in[5]: spatial shapes / level starts (compile-time constants here)
    const float* in_w   = (const float*)d_in[6];
    const float* in_b   = (const float*)d_in[7];
    const float* out_w  = (const float*)d_in[8];
    const float* out_b  = (const float*)d_in[9];
    const float* off_w  = (const float*)d_in[10];
    const float* off_b  = (const float*)d_in[11];
    const float* aw_w   = (const float*)d_in[12];
    const float* aw_b   = (const float*)d_in[13];
    const float* vp_w   = (const float*)d_in[14];
    const float* vp_b   = (const float*)d_in[15];
    const float* op_w   = (const float*)d_in[16];
    const float* op_b   = (const float*)d_in[17];
    const float* ln1_g  = (const float*)d_in[18];
    const float* ln1_b  = (const float*)d_in[19];
    const float* ln2_g  = (const float*)d_in[20];
    const float* ln2_b  = (const float*)d_in[21];
    const float* ln3_g  = (const float*)d_in[22];
    const float* ln3_b  = (const float*)d_in[23];
    const float* l1_w   = (const float*)d_in[24];
    const float* l1_b   = (const float*)d_in[25];
    const float* l2_w   = (const float*)d_in[26];
    const float* l2_b   = (const float*)d_in[27];

    float* w = (float*)d_ws;
    float* q_buf   = w;                              // 921600
    float* qkv     = q_buf  + (long)MTOK * D;        // 2764800
    float* tgt1    = qkv    + (long)MTOK * 3 * D;    // 921600
    float* x_buf   = tgt1   + (long)MTOK * D;        // 921600
    float* value   = x_buf  + (long)MTOK * D;        // 11523072
    float* off_buf = value  + (long)MMEM * D;        // 921600
    float* aw_buf  = off_buf+ (long)MTOK * D;        // 460800
    float* samp    = aw_buf + (long)MTOK * (D/2);    // 921600
    // aliases (lifetimes disjoint)
    float* attn    = q_buf;    // q_buf free after in_proj GEMMs
    float* sa_out  = qkv;      // qkv free after attention
    float* ca_out  = x_buf;    // x_buf free after off/aw GEMMs
    float* ffn_h   = value;    // value free after sampling
    float* ffn_out = samp + (long)MTOK * D;
    float* outp    = (float*)d_out;

    const int n_tok = MTOK * D;

    // --- self attention ---
    ew_add<<<(n_tok + 255) / 256, 256, 0, stream>>>(tgt, pos, q_buf, n_tok);
    launch_gemm(q_buf, D, in_w,             D, in_b,       qkv, 3 * D, 0,     MTOK, D, D, 0, stream);
    launch_gemm(q_buf, D, in_w + D * D,     D, in_b + D,   qkv, 3 * D, D,     MTOK, D, D, 0, stream);
    launch_gemm(tgt,   D, in_w + 2 * D * D, D, in_b + 2*D, qkv, 3 * D, 2 * D, MTOK, D, D, 0, stream);
    {
        dim3 g((NQ + 15) / 16, NH, BS);
        attn_flash<<<g, 32, 0, stream>>>(qkv, attn);
    }
    launch_gemm(attn, D, out_w, D, out_b, sa_out, D, 0, MTOK, D, D, 0, stream);
    resln<<<MTOK, 256, 0, stream>>>(tgt, sa_out, ln2_g, ln2_b, tgt1);

    // --- deformable cross attention ---
    ew_add<<<(n_tok + 255) / 256, 256, 0, stream>>>(tgt1, pos, x_buf, n_tok);
    launch_gemm(x_buf,  D, off_w, D, off_b, off_buf, D,     0, MTOK, D,     D, 0, stream);
    launch_gemm(x_buf,  D, aw_w,  D, aw_b,  aw_buf,  D / 2, 0, MTOK, D / 2, D, 0, stream);
    launch_gemm(memory, D, vp_w,  D, vp_b,  value,   D,     0, MMEM, D,     D, 0, stream);
    {
        dim3 g(NQ, BS);
        msdeform_sample<<<g, 256, 0, stream>>>(ref, value, off_buf, aw_buf, samp);
    }
    launch_gemm(samp, D, op_w, D, op_b, ca_out, D, 0, MTOK, D, D, 0, stream);
    resln<<<MTOK, 256, 0, stream>>>(tgt1, ca_out, ln1_g, ln1_b, tgt1);

    // --- FFN ---
    launch_gemm(tgt1,  D,    l1_w, D,    l1_b, ffn_h,   DFFN, 0, MTOK, DFFN, D,    1, stream);
    launch_gemm(ffn_h, DFFN, l2_w, DFFN, l2_b, ffn_out, D,    0, MTOK, D,    DFFN, 0, stream);
    resln<<<MTOK, 256, 0, stream>>>(tgt1, ffn_out, ln3_g, ln3_b, outp);
}
